// PolicyNetwork_49795850830260
// MI455X (gfx1250) — compile-verified, matrix operand order checked by reading, not run on hardware
//
#include <hip/hip_runtime.h>
#include <hip/hip_bf16.h>
#include <math.h>

#define HID       256
#define NNODES    100000
#define NEDGES    320000
#define NGRAPHS   2048
#define NLAYERS   5
#define NACT      18
#define LN_EPS    1e-5f

typedef __attribute__((ext_vector_type(16))) __bf16 v16bf;
typedef __attribute__((ext_vector_type(8)))  __bf16 v8bf;
typedef __attribute__((ext_vector_type(8)))  float  v8f;

// Build a 16-element bf16 WMMA fragment from two contiguous 8-element (16B)
// chunks at kA and kA+16 of a 256-wide row (CDNA5 16-bit A-fragment layout;
// B uses the mirrored layout because the weights are pre-transposed).
__device__ __forceinline__ v16bf make_frag(const __bf16* row, int kA) {
  v8bf lo = *(const v8bf*)(row + kA);
  v8bf hi = *(const v8bf*)(row + kA + 16);
  v16bf f;
#pragma unroll
  for (int i = 0; i < 8; ++i) { f[i] = lo[i]; f[i + 8] = hi[i]; }
  return f;
}

__device__ __forceinline__ float wave_sum(float v) {
#pragma unroll
  for (int m = 16; m > 0; m >>= 1) v += __shfl_xor(v, m, 32);
  return v;
}

// ---------- encoders: out = relu(X @ W + b), X:[M,K] K<=64, W:[K,256] ----------
__global__ void encoder_kernel(const float* __restrict__ X, const float* __restrict__ W,
                               const float* __restrict__ b, float* __restrict__ out, int K) {
  __shared__ float sx[64];
  const int m = blockIdx.x, c = threadIdx.x;
  if (c < K) sx[c] = X[(size_t)m * K + c];
  __syncthreads();
  float acc = b[c];
  for (int k = 0; k < K; ++k) acc = fmaf(sx[k], W[k * HID + c], acc);
  out[(size_t)m * HID + c] = fmaxf(acc, 0.f);
}

// ---------- transpose + convert all conv weights to bf16 (Bt[n][k] = W[k][n]) ----------
__global__ void wconv_kernel(const float* __restrict__ w1, const float* __restrict__ w2,
                             __bf16* __restrict__ w1t, __bf16* __restrict__ w2t) {
  const int idx = blockIdx.x * 256 + threadIdx.x;     // NLAYERS*256*256 total
  const int l = idx >> 16, r = idx & 65535;
  const int k = r >> 8, n = r & 255;
  w1t[l * 65536 + n * 256 + k] = (__bf16)w1[idx];
  w2t[l * 65536 + n * 256 + k] = (__bf16)w2[idx];
}

__global__ void zero_kernel(float* __restrict__ p, long n) {
  const long i = (long)blockIdx.x * blockDim.x + threadIdx.x;
  if (i < n) p[i] = 0.f;
}

// ---------- edge message + scatter-add: agg[dst] += relu(h[src] + e) ----------
__global__ void scatter_kernel(const float* __restrict__ h, const float* __restrict__ e,
                               const int* __restrict__ ei, float* __restrict__ agg) {
  const int eid = blockIdx.x, c = threadIdx.x;
  const int src = ei[eid];
  const int dst = ei[NEDGES + eid];
  float v = h[(size_t)src * HID + c] + e[(size_t)eid * HID + c];
  v = fmaxf(v, 0.f);
  atomicAdd(&agg[(size_t)dst * HID + c], v);
}

// ===================== WMMA GEMM #1 =====================
// y1 = relu(Z @ W1^T(bf16) + b1), Z is f32 [M,256]; f32->bf16 conversion is
// fused into the LDS staging. 8 waves/block, wave w owns columns [32w,32w+32).
// Full bf16 B (256x256, pre-transposed) register-resident; grid-stride over
// row tiles for B reuse.
__global__ __launch_bounds__(256) void gemm1_kernel(
    const float* __restrict__ A, const __bf16* __restrict__ Bt,
    const float* __restrict__ bias, __bf16* __restrict__ out, int M) {
  __shared__ __bf16 sA[16][264];                       // +16B row pad vs LDS banks
  const int lane = threadIdx.x & 31;
  const int wave = threadIdx.x >> 5;
  const int hlf  = lane >> 4;
  const int l16  = lane & 15;

  v16bf bfr[2][8];
#pragma unroll
  for (int ct = 0; ct < 2; ++ct) {
    const __bf16* brow = Bt + (size_t)(wave * 32 + ct * 16 + l16) * 256;
#pragma unroll
    for (int k = 0; k < 8; ++k)
      bfr[ct][k] = make_frag(brow, k * 32 + hlf * 8);
  }

  const int tiles = M >> 4;
  for (int t = blockIdx.x; t < tiles; t += gridDim.x) {
    const float* Atile = A + (size_t)t * (16 * 256);
    if (t + (int)gridDim.x < tiles)
      __builtin_prefetch(A + (size_t)(t + gridDim.x) * (16 * 256), 0, 3);
    // Stage 16x256 f32 tile -> bf16 LDS tile (convert on the fly).
    for (int i = threadIdx.x; i < 1024; i += 256) {
      const int r = i >> 6, cc = i & 63;
      const float4 f = *(const float4*)(Atile + r * 256 + cc * 4);
      __bf16* d = &sA[r][cc * 4];
      d[0] = (__bf16)f.x; d[1] = (__bf16)f.y; d[2] = (__bf16)f.z; d[3] = (__bf16)f.w;
    }
    __syncthreads();

    v8f acc0 = {}; v8f acc1 = {};
#pragma unroll
    for (int k = 0; k < 8; ++k) {
      v16bf af = make_frag(&sA[l16][0], k * 32 + hlf * 8);
      acc0 = __builtin_amdgcn_wmma_f32_16x16x32_bf16(false, af, false, bfr[0][k],
                                                     (short)0, acc0, false, false);
      acc1 = __builtin_amdgcn_wmma_f32_16x16x32_bf16(false, af, false, bfr[1][k],
                                                     (short)0, acc1, false, false);
    }
    __syncthreads();

    const int n0 = wave * 32 + l16;
#pragma unroll
    for (int v = 0; v < 8; ++v) {
      const size_t m = (size_t)t * 16 + v + hlf * 8;
      out[m * HID + n0]      = (__bf16)fmaxf(acc0[v] + bias[n0], 0.f);
      out[m * HID + n0 + 16] = (__bf16)fmaxf(acc1[v] + bias[n0 + 16], 0.f);
    }
  }
}

// ===================== WMMA GEMM #2 + fused LN epilogue =====================
// y = Y1(bf16) @ W2^T + b2; then h = LN(relu(LN(y;cg,cb)) + h; lg,lb), in place.
// The block covers the full 256-wide rows of its 16-row tile, so the row-wise
// LayerNorms run entirely in-block: accumulators are parked in a padded f32
// LDS tile and one wave32 handles two rows with shuffle reductions.
__global__ __launch_bounds__(256) void gemm2_ln_kernel(
    const __bf16* __restrict__ A, const __bf16* __restrict__ Bt,
    const float* __restrict__ bias,
    const float* __restrict__ cg, const float* __restrict__ cb,
    const float* __restrict__ lg, const float* __restrict__ lb,
    float* __restrict__ h, int M) {
  __shared__ __bf16 sA[16][264];
  __shared__ float  sO[16][257];                       // +1 pad vs LDS banks
  const int lane = threadIdx.x & 31;
  const int wave = threadIdx.x >> 5;
  const int hlf  = lane >> 4;
  const int l16  = lane & 15;

  v16bf bfr[2][8];
#pragma unroll
  for (int ct = 0; ct < 2; ++ct) {
    const __bf16* brow = Bt + (size_t)(wave * 32 + ct * 16 + l16) * 256;
#pragma unroll
    for (int k = 0; k < 8; ++k)
      bfr[ct][k] = make_frag(brow, k * 32 + hlf * 8);
  }

  const int tiles = M >> 4;
  for (int t = blockIdx.x; t < tiles; t += gridDim.x) {
    const __bf16* Atile = A + (size_t)t * (16 * 256);
    if (t + (int)gridDim.x < tiles)
      __builtin_prefetch(A + (size_t)(t + gridDim.x) * (16 * 256), 0, 3);
    for (int i = threadIdx.x; i < 512; i += 256) {
      const int r = i >> 5, cc = i & 31;
      *(uint4*)&sA[r][cc * 8] = *(const uint4*)(Atile + r * 256 + cc * 8);
    }
    __syncthreads();

    v8f acc0 = {}; v8f acc1 = {};
#pragma unroll
    for (int k = 0; k < 8; ++k) {
      v16bf af = make_frag(&sA[l16][0], k * 32 + hlf * 8);
      acc0 = __builtin_amdgcn_wmma_f32_16x16x32_bf16(false, af, false, bfr[0][k],
                                                     (short)0, acc0, false, false);
      acc1 = __builtin_amdgcn_wmma_f32_16x16x32_bf16(false, af, false, bfr[1][k],
                                                     (short)0, acc1, false, false);
    }

    // Park accumulators (+bias) in LDS: row-major 16x256 f32 tile.
    const int n0 = wave * 32 + l16;
#pragma unroll
    for (int v = 0; v < 8; ++v) {
      const int ml = v + hlf * 8;                      // D layout: M=v (+8 hi half)
      sO[ml][n0]      = acc0[v] + bias[n0];
      sO[ml][n0 + 16] = acc1[v] + bias[n0 + 16];
    }
    __syncthreads();

    // Fused epilogue: one wave per 2 rows.
#pragma unroll
    for (int rr = 0; rr < 2; ++rr) {
      const int r = wave * 2 + rr;
      float* hr = h + ((size_t)t * 16 + r) * HID;
      float v[8];
#pragma unroll
      for (int j = 0; j < 8; ++j) v[j] = sO[r][lane + j * 32];
      float s = 0.f;
#pragma unroll
      for (int j = 0; j < 8; ++j) s += v[j];
      const float mu = wave_sum(s) * (1.f / HID);
      float s2 = 0.f;
#pragma unroll
      for (int j = 0; j < 8; ++j) { const float d = v[j] - mu; s2 += d * d; }
      const float inv = rsqrtf(wave_sum(s2) * (1.f / HID) + LN_EPS);
      float tv[8];
#pragma unroll
      for (int j = 0; j < 8; ++j) {
        const int c = lane + j * 32;
        float z = (v[j] - mu) * inv * cg[c] + cb[c];
        z = fmaxf(z, 0.f);
        tv[j] = z + hr[c];                             // residual
      }
      s = 0.f;
#pragma unroll
      for (int j = 0; j < 8; ++j) s += tv[j];
      const float mu2 = wave_sum(s) * (1.f / HID);
      s2 = 0.f;
#pragma unroll
      for (int j = 0; j < 8; ++j) { const float d = tv[j] - mu2; s2 += d * d; }
      const float inv2 = rsqrtf(wave_sum(s2) * (1.f / HID) + LN_EPS);
#pragma unroll
      for (int j = 0; j < 8; ++j) {
        const int c = lane + j * 32;
        hr[c] = (tv[j] - mu2) * inv2 * lg[c] + lb[c];
      }
    }
    __syncthreads();
  }
}

// ---------- per-graph pooling ----------
__global__ void pool_kernel(const float* __restrict__ h, const int* __restrict__ batch,
                            float* __restrict__ sums, float* __restrict__ counts) {
  const int n = blockIdx.x, c = threadIdx.x;
  const int g = batch[n];
  atomicAdd(&sums[(size_t)g * HID + c], h[(size_t)n * HID + c]);
  if (c == 0) atomicAdd(&counts[g], 1.f);
}

__global__ void gbuild_kernel(const float* __restrict__ sums, const float* __restrict__ counts,
                              float* __restrict__ g) {
  const int idx = blockIdx.x * 256 + threadIdx.x;      // NGRAPHS*512
  const int gi = idx >> 9, c = idx & 511;
  float s;
  if (c < HID) {
    s = sums[gi * HID + c] / fmaxf(counts[gi], 1.f);   // means
  } else {
    s = sums[gi * HID + (c - HID)];                    // sums
  }
  g[idx] = s;
}

// ---------- small dense head ----------
__global__ void dense_kernel(const float* __restrict__ A, const float* __restrict__ W,
                             const float* __restrict__ b, float* __restrict__ out,
                             int K, int N, int relu) {
  const int m = blockIdx.x, n = threadIdx.x;
  if (n >= N) return;
  float acc = b[n];
  const float* ar = A + (size_t)m * K;
  for (int k = 0; k < K; ++k) acc = fmaf(ar[k], W[k * N + n], acc);
  if (relu) acc = fmaxf(acc, 0.f);
  out[(size_t)m * N + n] = acc;
}

__global__ void softmax_kernel(const float* __restrict__ logits, float* __restrict__ out) {
  const int g = blockIdx.x * blockDim.x + threadIdx.x;
  if (g >= NGRAPHS) return;
  const float* lr = logits + g * NACT;
  float mx = lr[0];
  for (int a = 1; a < NACT; ++a) mx = fmaxf(mx, lr[a]);
  float se = 0.f;
  for (int a = 0; a < NACT; ++a) se += __expf(lr[a] - mx);
  const float lse = __logf(se);
  for (int a = 0; a < NACT; ++a) {
    const float lsm = lr[a] - mx - lse;
    out[g * NACT + a] = lsm;                           // log_softmax
    out[NGRAPHS * NACT + g * NACT + a] = __expf(lsm);  // softmax
  }
}

extern "C" void kernel_launch(void* const* d_in, const int* in_sizes, int n_in,
                              void* d_out, int out_size, void* d_ws, size_t ws_size,
                              hipStream_t stream) {
  (void)in_sizes; (void)n_in; (void)out_size; (void)ws_size;
  const float* x       = (const float*)d_in[0];
  const int*   ei      = (const int*)d_in[1];
  const float* eattr   = (const float*)d_in[2];
  const int*   batch   = (const int*)d_in[3];
  const float* node_w  = (const float*)d_in[4];
  const float* node_b  = (const float*)d_in[5];
  const float* edge_w  = (const float*)d_in[6];
  const float* edge_b  = (const float*)d_in[7];
  const float* conv_w1 = (const float*)d_in[8];
  const float* conv_b1 = (const float*)d_in[9];
  const float* conv_w2 = (const float*)d_in[10];
  const float* conv_b2 = (const float*)d_in[11];
  const float* cnorm_g = (const float*)d_in[12];
  const float* cnorm_b = (const float*)d_in[13];
  const float* lnorm_g = (const float*)d_in[14];
  const float* lnorm_b = (const float*)d_in[15];
  const float* fp_w1   = (const float*)d_in[16];
  const float* fp_b1   = (const float*)d_in[17];
  const float* fp_w2   = (const float*)d_in[18];
  const float* fp_b2   = (const float*)d_in[19];
  const float* fp_w3   = (const float*)d_in[20];
  const float* fp_b3   = (const float*)d_in[21];

  char* ws = (char*)d_ws;
  size_t off = 0;
  auto take = [&](size_t bytes) -> char* {
    char* p = ws + off;
    off += (bytes + 255) & ~(size_t)255;
    return p;
  };
  float*  h    = (float*)take((size_t)NNODES * HID * 4);
  float*  e    = (float*)take((size_t)NEDGES * HID * 4);
  float*  agg  = (float*)take((size_t)NNODES * HID * 4);
  __bf16* y1b  = (__bf16*)take((size_t)NNODES * HID * 2);
  __bf16* w1t  = (__bf16*)take((size_t)NLAYERS * HID * HID * 2);
  __bf16* w2t  = (__bf16*)take((size_t)NLAYERS * HID * HID * 2);
  float*  sums = (float*)take((size_t)NGRAPHS * HID * 4);
  float*  cnts = (float*)take((size_t)NGRAPHS * 4);
  float*  gbuf = (float*)take((size_t)NGRAPHS * 2 * HID * 4);
  float*  h1   = (float*)take((size_t)NGRAPHS * HID * 4);
  float*  h2   = (float*)take((size_t)NGRAPHS * (HID / 2) * 4);
  float*  lgt  = (float*)take((size_t)NGRAPHS * NACT * 4);

  // encoders + one-time weight convert/transpose to bf16
  encoder_kernel<<<NNODES, HID, 0, stream>>>(x, node_w, node_b, h, 38);
  encoder_kernel<<<NEDGES, HID, 0, stream>>>(eattr, edge_w, edge_b, e, 7);
  wconv_kernel<<<(NLAYERS * HID * HID) / 256, 256, 0, stream>>>(conv_w1, conv_w2, w1t, w2t);

  for (int l = 0; l < NLAYERS; ++l) {
    // agg = h, then agg[dst] += relu(h[src] + e)   (z = h + segment_sum(msg))
    hipMemcpyAsync(agg, h, (size_t)NNODES * HID * 4, hipMemcpyDeviceToDevice, stream);
    scatter_kernel<<<NEDGES, HID, 0, stream>>>(h, e, ei, agg);
    // y1 = relu(z @ W1 + b1)  (f32 in, bf16 out; convert fused in staging)
    gemm1_kernel<<<250, 256, 0, stream>>>(agg, w1t + (size_t)l * HID * HID,
                                          conv_b1 + l * HID, y1b, NNODES);
    // h = LN(relu(LN(y1 @ W2 + b2)) + h)  (fully fused epilogue)
    gemm2_ln_kernel<<<250, 256, 0, stream>>>(y1b, w2t + (size_t)l * HID * HID,
                                             conv_b2 + l * HID,
                                             cnorm_g + l * HID, cnorm_b + l * HID,
                                             lnorm_g + l * HID, lnorm_b + l * HID,
                                             h, NNODES);
  }

  // pooling + head
  zero_kernel<<<(NGRAPHS * HID) / 256, 256, 0, stream>>>(sums, (long)NGRAPHS * HID);
  zero_kernel<<<(NGRAPHS + 255) / 256, 256, 0, stream>>>(cnts, (long)NGRAPHS);
  pool_kernel<<<NNODES, HID, 0, stream>>>(h, batch, sums, cnts);
  gbuild_kernel<<<(NGRAPHS * 2 * HID) / 256, 256, 0, stream>>>(sums, cnts, gbuf);
  dense_kernel<<<NGRAPHS, 256, 0, stream>>>(gbuf, fp_w1, fp_b1, h1, 2 * HID, HID, 1);
  dense_kernel<<<NGRAPHS, 256, 0, stream>>>(h1, fp_w2, fp_b2, h2, HID, HID / 2, 1);
  dense_kernel<<<NGRAPHS, 256, 0, stream>>>(h2, fp_w3, fp_b3, lgt, HID / 2, NACT, 0);
  softmax_kernel<<<(NGRAPHS + 255) / 256, 256, 0, stream>>>(lgt, (float*)d_out);
}
